// Mamba_TS_69595650064652
// MI455X (gfx1250) — matrix operand order repaired and checked by reading, not compiled, not run on hardware
//
#include <hip/hip_runtime.h>
#include <hip/hip_bf16.h>
#include <math.h>

typedef __attribute__((ext_vector_type(16))) _Float16 v16h;
typedef __attribute__((ext_vector_type(8)))  float    v8f;

static constexpr int BATCH = 16;
static constexpr int SEQ   = 2048;
static constexpr int FEAT  = 32;
static constexpr int Hdim  = 256;
static constexpr int Ndim  = 16;
static constexpr int Kconv = 4;
static constexpr int Lnum  = 4;
static constexpr int HOR   = 24;
static constexpr int TOK   = BATCH * SEQ;   // 32768 tokens

// ---------------------------------------------------------------------------
// Generic WMMA GEMM:  C[M x Nout] = act( A[M x Kd] * W[Nout x Kd]^T + bias )
// f32 inputs, staged to LDS as f16, accumulated with v_wmma_f32_16x16x32_f16.
// Block = 128 threads = 4 waves; block tile = 64(M) x 16*NW(N); K-step = 32.
// Each wave computes one 16x(16*NW) strip: NW WMMAs sharing one A fragment.
// Requires: M % 64 == 0, Kd % 32 == 0, Nout % (16*NW) == 0.
// act: 0 = none, 1 = relu, 2 = softplus
// ---------------------------------------------------------------------------
template <int NW>
__global__ __launch_bounds__(128) void mamba_gemm_wmma(
    const float* __restrict__ A, const float* __restrict__ W,
    const float* __restrict__ bias, float* __restrict__ C,
    int Kd, int Nout, int act)
{
    __shared__ _Float16 As[64][40];        // padded rows: 80B, 16B aligned
    __shared__ _Float16 Ws[16 * NW][40];

    const int tid  = threadIdx.x;
    const int wv   = tid >> 5;             // wave id 0..3 (wave32)
    const int lane = tid & 31;
    const int m16  = lane & 15;
    const bool hi  = lane >= 16;

    const long rowBase = (long)blockIdx.x * 64;
    const int  colBase = blockIdx.y * (16 * NW);

    v8f acc[NW];
    #pragma unroll
    for (int j = 0; j < NW; ++j) acc[j] = (v8f){};

    for (int k0 = 0; k0 < Kd; k0 += 32) {
        // --- cooperative load A tile (64 x 32 f32 -> f16) ---
        {
            const int r  = tid >> 1;
            const int sg = (tid & 1) * 16;
            const float* src = A + (rowBase + r) * (long)Kd + k0 + sg;
            if (k0 + 32 < Kd) __builtin_prefetch(src + 32, 0, 1);
            #pragma unroll
            for (int j = 0; j < 16; ++j) As[r][sg + j] = (_Float16)src[j];
        }
        // --- cooperative load W tile (16*NW x 32 f32 -> f16) ---
        {
            const int per = NW * 4;        // NW*512 elements / 128 threads
            const int idx = tid * per;
            #pragma unroll
            for (int j = 0; j < per; ++j) {
                const int e  = idx + j;
                const int r  = e >> 5;
                const int cc = e & 31;
                Ws[r][cc] = (_Float16)W[(colBase + r) * (long)Kd + k0 + cc];
            }
        }
        __syncthreads();

        // --- A fragment per CDNA5 16x16x32 f16 layout (shared by NW WMMAs) ---
        v16h av;
        const int aRow = wv * 16 + m16;
        const int kA0 = hi ? 8 : 0, kA1 = hi ? 24 : 16;
        #pragma unroll
        for (int j = 0; j < 8; ++j) {
            av[j]     = As[aRow][kA0 + j];
            av[8 + j] = As[aRow][kA1 + j];
        }
        const int kB0 = hi ? 8 : 0, kB1 = hi ? 24 : 16;
        #pragma unroll
        for (int t = 0; t < NW; ++t) {
            v16h bv;
            #pragma unroll
            for (int j = 0; j < 8; ++j) {
                bv[j]     = Ws[t * 16 + m16][kB0 + j];
                bv[8 + j] = Ws[t * 16 + m16][kB1 + j];
            }
            acc[t] = __builtin_amdgcn_wmma_f32_16x16x32_f16(
                /*neg_a=*/false, av, /*neg_b=*/false, bv,
                /*c_mod=*/(short)0, acc[t], /*reuse_a=*/false, /*reuse_b=*/false);
        }
        __syncthreads();
    }

    // --- epilogue: bias + activation, C/D VGPR layout ---
    #pragma unroll
    for (int t = 0; t < NW; ++t) {
        #pragma unroll
        for (int r = 0; r < 8; ++r) {
            const int  m   = r + (hi ? 8 : 0);
            const long row = rowBase + wv * 16 + m;
            const int  col = colBase + t * 16 + m16;
            float v = acc[t][r] + bias[col];
            if (act == 1)      v = v > 0.f ? v : 0.f;
            else if (act == 2) v = (v > 20.f) ? v : __logf(1.f + __expf(v));
            C[row * (long)Nout + col] = v;
        }
    }
}

// ---------------------------------------------------------------------------
// Causal depthwise conv (K=4) + SiLU.  xi rows are 512 wide; x_ssm = xi[:,0:256].
// ---------------------------------------------------------------------------
__global__ __launch_bounds__(256) void mamba_conv_silu(
    const float* __restrict__ xi, const float* __restrict__ cw,
    const float* __restrict__ cb, float* __restrict__ xc)
{
    const long idx = (long)blockIdx.x * 256 + threadIdx.x;   // TOK*H threads
    const int  h   = (int)(idx & (Hdim - 1));
    const long T   = idx >> 8;                               // token index
    const int  s   = (int)(T & (SEQ - 1));
    const long bT0 = T - s;                                  // batch row base

    float acc = cb[h];
    #pragma unroll
    for (int k = 0; k < Kconv; ++k) {
        const int sp = s - (Kconv - 1) + k;
        if (sp >= 0)
            acc += xi[(bT0 + sp) * 512 + h] * cw[h * Kconv + k];
    }
    const float sig = 1.f / (1.f + __expf(-acc));
    xc[T * Hdim + h] = acc * sig;
}

// ---------------------------------------------------------------------------
// Sequential selective scan.  Grid (H/16, B), block 256 = 16 h-channels x 16 n.
// Reduction over N via in-wave shfl_xor (width 16, wave32).
// Fuses y *= silu(gate), gate = xi[:, 256+h].
// ---------------------------------------------------------------------------
__global__ __launch_bounds__(256) void mamba_ssm_scan(
    const float* __restrict__ xc, const float* __restrict__ dt,
    const float* __restrict__ bc, const float* __restrict__ Alog,
    const float* __restrict__ Dp, const float* __restrict__ xi,
    float* __restrict__ y)
{
    const int n  = threadIdx.x & 15;
    const int hl = threadIdx.x >> 4;
    const int h  = blockIdx.x * 16 + hl;
    const int b  = blockIdx.y;

    const float Am = -__expf(Alog[h * Ndim + n]);
    const float Dv = Dp[h];
    float s = 0.f;

    const long tok0 = (long)b * SEQ;
    for (int t = 0; t < SEQ; ++t) {
        const long tok = tok0 + t;
        const float x  = xc[tok * Hdim + h];
        const float d  = dt[tok * Hdim + h];
        const float Bv = bc[tok * 32 + n];
        const float Cv = bc[tok * 32 + Ndim + n];

        s = s * __expf(d * Am) + (d * x) * Bv;

        float contrib = s * Cv;
        #pragma unroll
        for (int off = 8; off > 0; off >>= 1)
            contrib += __shfl_xor(contrib, off, 16);

        if (n == 0) {
            const float g   = xi[tok * 512 + Hdim + h];
            const float sig = 1.f / (1.f + __expf(-g));
            y[tok * Hdim + h] = (contrib + x * Dv) * (g * sig);
        }
    }
}

// ---------------------------------------------------------------------------
// Residual + LayerNorm over H=256.  One wave per row, 8 rows per 256-thr block.
// ---------------------------------------------------------------------------
__global__ __launch_bounds__(256) void mamba_residual_ln(
    const float* __restrict__ tmp, const float* __restrict__ hin,
    const float* __restrict__ g, const float* __restrict__ bb,
    float* __restrict__ hout)
{
    const int wv   = threadIdx.x >> 5;
    const int lane = threadIdx.x & 31;
    const long row = (long)blockIdx.x * 8 + wv;
    const long base = row * Hdim;

    float v[8], sum = 0.f, sq = 0.f;
    #pragma unroll
    for (int j = 0; j < 8; ++j) {
        const int i = lane + j * 32;
        v[j] = tmp[base + i] + hin[base + i];
        sum += v[j];
        sq  += v[j] * v[j];
    }
    #pragma unroll
    for (int off = 16; off > 0; off >>= 1) {
        sum += __shfl_xor(sum, off, 32);
        sq  += __shfl_xor(sq,  off, 32);
    }
    const float mu  = sum * (1.f / Hdim);
    const float var = sq * (1.f / Hdim) - mu * mu;
    const float rs  = rsqrtf(var + 1e-5f);
    #pragma unroll
    for (int j = 0; j < 8; ++j) {
        const int i = lane + j * 32;
        hout[base + i] = (v[j] - mu) * rs * g[i] + bb[i];
    }
}

// ---------------------------------------------------------------------------
// Prediction head on last token: [16 x 256] -> relu(128) -> [16 x 24].
// One block per batch element.
// ---------------------------------------------------------------------------
__global__ __launch_bounds__(128) void mamba_pred_head(
    const float* __restrict__ hfin,
    const float* __restrict__ pw1, const float* __restrict__ pb1,
    const float* __restrict__ pw2, const float* __restrict__ pb2,
    float* __restrict__ pred)
{
    __shared__ float mid[Hdim / 2];
    const int b = blockIdx.x;
    const int j = threadIdx.x;
    const float* last = hfin + ((long)b * SEQ + (SEQ - 1)) * Hdim;

    float acc = pb1[j];
    for (int i = 0; i < Hdim; ++i) acc += last[i] * pw1[j * Hdim + i];
    mid[j] = acc > 0.f ? acc : 0.f;
    __syncthreads();

    if (j < HOR) {
        float o = pb2[j];
        #pragma unroll 4
        for (int k = 0; k < Hdim / 2; ++k) o += mid[k] * pw2[j * (Hdim / 2) + k];
        pred[b * HOR + j] = o;
    }
}

// ---------------------------------------------------------------------------
extern "C" void kernel_launch(void* const* d_in, const int* in_sizes, int n_in,
                              void* d_out, int out_size, void* d_ws, size_t ws_size,
                              hipStream_t stream)
{
    (void)in_sizes; (void)n_in; (void)out_size; (void)ws_size;

    const float* x    = (const float*)d_in[0];
    const float* ipw  = (const float*)d_in[1];
    const float* ipb  = (const float*)d_in[2];
    const float* inw  = (const float*)d_in[3];   // (L,512,256)
    const float* inb  = (const float*)d_in[4];   // (L,512)
    const float* cw   = (const float*)d_in[5];   // (L,256,4)
    const float* cb   = (const float*)d_in[6];   // (L,256)
    const float* bcw  = (const float*)d_in[7];   // (L,32,256)
    const float* bcb  = (const float*)d_in[8];   // (L,32)
    const float* dtw  = (const float*)d_in[9];   // (L,256,256)
    const float* dtb  = (const float*)d_in[10];  // (L,256)
    const float* Alog = (const float*)d_in[11];  // (L,256,16)
    const float* Dp   = (const float*)d_in[12];  // (L,256)
    const float* ow   = (const float*)d_in[13];  // (L,256,256)
    const float* ob   = (const float*)d_in[14];  // (L,256)
    const float* lng  = (const float*)d_in[15];  // (L,256)
    const float* lnb  = (const float*)d_in[16];  // (L,256)
    const float* rw1  = (const float*)d_in[17];
    const float* rb1  = (const float*)d_in[18];
    const float* rw2  = (const float*)d_in[19];
    const float* rb2  = (const float*)d_in[20];
    const float* pw1  = (const float*)d_in[21];
    const float* pb1  = (const float*)d_in[22];
    const float* pw2  = (const float*)d_in[23];
    const float* pb2  = (const float*)d_in[24];

    float* out = (float*)d_out;
    const long reconN = (long)TOK * FEAT;        // 1,048,576

    // workspace carving (floats)
    float* ws = (float*)d_ws;
    size_t off = 0;
    float* hA  = ws + off; off += (size_t)TOK * Hdim;        // 32 MB
    float* hB  = ws + off; off += (size_t)TOK * Hdim;        // 32 MB
    float* xi  = ws + off; off += (size_t)TOK * 2 * Hdim;    // 64 MB
    float* xc  = ws + off; off += (size_t)TOK * Hdim;        // 32 MB
    float* dtv = ws + off; off += (size_t)TOK * Hdim;        // 32 MB
    float* bcv = ws + off; off += (size_t)TOK * 32;          //  4 MB
    float* yv  = ws + off; off += (size_t)TOK * Hdim;        // 32 MB
    float* tmp = xi;                                         // reuse after gate consumed
    float* mid = xc;                                         // reuse for recon hidden

    const dim3 gemmBlk(128);
    #define GRID4(NOUT) dim3(TOK / 64, (NOUT) / 64)
    #define GRID2(NOUT) dim3(TOK / 64, (NOUT) / 32)

    // 1) input projection: x[32768x32] -> h[32768x256]
    hipLaunchKernelGGL(mamba_gemm_wmma<4>, GRID4(Hdim), gemmBlk, 0, stream,
                       x, ipw, ipb, hA, FEAT, Hdim, 0);

    float* hcur = hA;
    float* hnxt = hB;
    for (int l = 0; l < Lnum; ++l) {
        // a) in_proj: h -> xi [32768x512]
        hipLaunchKernelGGL(mamba_gemm_wmma<4>, GRID4(2 * Hdim), gemmBlk, 0, stream,
                           hcur, inw + (size_t)l * 512 * Hdim, inb + (size_t)l * 512,
                           xi, Hdim, 2 * Hdim, 0);
        // b) causal dwconv + silu -> xc
        hipLaunchKernelGGL(mamba_conv_silu, dim3(TOK * Hdim / 256), dim3(256), 0, stream,
                           xi, cw + (size_t)l * Hdim * Kconv, cb + (size_t)l * Hdim, xc);
        // c) BC projection: xc -> bc [32768x32]
        hipLaunchKernelGGL(mamba_gemm_wmma<2>, GRID2(32), gemmBlk, 0, stream,
                           xc, bcw + (size_t)l * 32 * Hdim, bcb + (size_t)l * 32,
                           bcv, Hdim, 32, 0);
        // d) dt projection + softplus: xc -> dt [32768x256]
        hipLaunchKernelGGL(mamba_gemm_wmma<4>, GRID4(Hdim), gemmBlk, 0, stream,
                           xc, dtw + (size_t)l * Hdim * Hdim, dtb + (size_t)l * Hdim,
                           dtv, Hdim, Hdim, 2);
        // e) sequential scan + gate silu -> y
        hipLaunchKernelGGL(mamba_ssm_scan, dim3(Hdim / 16, BATCH), dim3(256), 0, stream,
                           xc, dtv, bcv, Alog + (size_t)l * Hdim * Ndim,
                           Dp + (size_t)l * Hdim, xi, yv);
        // f) out_proj: y -> tmp (reuses xi)
        hipLaunchKernelGGL(mamba_gemm_wmma<4>, GRID4(Hdim), gemmBlk, 0, stream,
                           yv, ow + (size_t)l * Hdim * Hdim, ob + (size_t)l * Hdim,
                           tmp, Hdim, Hdim, 0);
        // g) residual + layernorm -> hnxt
        hipLaunchKernelGGL(mamba_residual_ln, dim3(TOK / 8), dim3(256), 0, stream,
                           tmp, hcur, lng + (size_t)l * Hdim, lnb + (size_t)l * Hdim, hnxt);
        float* t = hcur; hcur = hnxt; hnxt = t;
    }

    // 3) recon head: h -> relu(128) -> 32, second GEMM writes straight into d_out
    hipLaunchKernelGGL(mamba_gemm_wmma<4>, GRID4(Hdim / 2), gemmBlk, 0, stream,
                       hcur, rw1, rb1, mid, Hdim, Hdim / 2, 1);
    hipLaunchKernelGGL(mamba_gemm_wmma<2>, GRID2(FEAT), gemmBlk, 0, stream,
                       mid, rw2, rb2, out, Hdim / 2, FEAT, 0);

    // 4) pred head on last token -> out[reconN ..]
    hipLaunchKernelGGL(mamba_pred_head, dim3(BATCH), dim3(128), 0, stream,
                       hcur, pw1, pb1, pw2, pb2, out + reconN);

    #undef GRID4
    #undef GRID2
}
